// FlashGPT2Attention_51565377356299
// MI455X (gfx1250) — matrix operand order repaired and checked by reading, compiled, not run on hardware
//
#include <hip/hip_runtime.h>
#include <hip/hip_bf16.h>
#include <stdint.h>

// ---------------------------------------------------------------------------
// Types for CDNA5 WMMA (wave32): v16bf A/B fragments, v8f C/D accumulators.
// ---------------------------------------------------------------------------
typedef __attribute__((ext_vector_type(8)))  float          v8f;
typedef __attribute__((ext_vector_type(16))) unsigned short v16us;
typedef __attribute__((ext_vector_type(8)))  unsigned short v8us;
typedef __attribute__((ext_vector_type(16))) __bf16         v16bf;

union BfCast { v16us u; v16bf b; };
union Frag   { v16us v; v8us  h[2]; };

__device__ __forceinline__ unsigned short f32_to_bf16(float f) {
  unsigned int u = __float_as_uint(f);
  u += 0x7FFFu + ((u >> 16) & 1u);      // round-to-nearest-even
  return (unsigned short)(u >> 16);
}

__device__ __forceinline__ v8f wmma_bf16(v16us a, v16us b, v8f c) {
  BfCast A, B; A.u = a; B.u = b;
  return __builtin_amdgcn_wmma_f32_16x16x32_bf16(
      /*neg_a=*/false, A.b, /*neg_b=*/false, B.b,
      /*c_mod=*/(short)0, c, /*reuse_a=*/false, /*reuse_b=*/false);
}

// ---------------------------------------------------------------------------
// Async global->LDS b128 copies (CDNA5 GLOBAL_LOAD_ASYNC_TO_LDS_B128,
// tracked by ASYNCcnt). Parameter type per hipcc diagnostic: pointer to
// GCC-vector 'int4' (generic address space). Falls back to synchronous
// uint4 copies if the builtin is not declared (and on the host pass).
// ---------------------------------------------------------------------------
#if defined(__has_builtin)
#  if __has_builtin(__builtin_amdgcn_global_load_async_to_lds_b128)
#    define HAVE_ASYNC_LDS 1
#  endif
#endif
#ifndef HAVE_ASYNC_LDS
#  define HAVE_ASYNC_LDS 0
#endif

#if HAVE_ASYNC_LDS
typedef int async_v4i __attribute__((vector_size(4 * sizeof(int))));
__device__ __forceinline__ void async_copy_b128(const void* g, void* l) {
  __builtin_amdgcn_global_load_async_to_lds_b128(
      (async_v4i*)g, (async_v4i*)l, 0, 0);
}
__device__ __forceinline__ void async_wait() {
#if __has_builtin(__builtin_amdgcn_s_wait_asynccnt)
  __builtin_amdgcn_s_wait_asynccnt(0);
#else
  asm volatile("s_wait_asynccnt 0x0" ::: "memory");
#endif
}
#else
__device__ __forceinline__ void async_copy_b128(const void* g, void* l) {
  *(uint4*)l = *(const uint4*)g;
}
__device__ __forceinline__ void async_wait() {}
#endif

// Problem constants
#define TT   4096
#define HID  2048
#define NH   16
#define HD   128
#define NB   4
#define SS   1024
#define NSLOT 8192

// ---------------------------------------------------------------------------
// Zero-fill (cache regions of d_out must start at 0.0f; harness poisons).
// ---------------------------------------------------------------------------
__global__ void zero_kernel(float* __restrict__ p, long n4) {
  long i = (long)blockIdx.x * blockDim.x + threadIdx.x;
  long stride = (long)gridDim.x * blockDim.x;
  float4* p4 = (float4*)p;
  float4 z = make_float4(0.f, 0.f, 0.f, 0.f);
  for (long j = i; j < n4; j += stride) p4[j] = z;
}

// ---------------------------------------------------------------------------
// QKV GEMM: C[4096,6144] = hs[4096,2048] @ W[2048,6144] + bias
// f32 -> bf16 conversion while staging to LDS; bf16 WMMA, f32 accumulate.
// Epilogue routes Q/K/V to bf16 workspaces + f32 cache scatter via slots.
// ---------------------------------------------------------------------------
__global__ __launch_bounds__(256) void qkv_gemm_kernel(
    const float* __restrict__ A, const float* __restrict__ Bw,
    const float* __restrict__ bias, const int* __restrict__ slots,
    unsigned short* __restrict__ wq, unsigned short* __restrict__ wk,
    unsigned short* __restrict__ wv,
    float* __restrict__ kcache, float* __restrict__ vcache)
{
  const int K = HID, N = 3 * HID;
  __shared__ __align__(64) unsigned short lA[128][32];   // [m][k] bf16
  __shared__ __align__(64) unsigned short lB[128][32];   // [n][k] bf16 (transposed)

  const int tid  = threadIdx.x;
  const int lane = tid & 31, wave = tid >> 5;
  const int wm = wave >> 1, wn = wave & 1;               // 4x2 wave grid
  const int half = lane >> 4, lm = lane & 15;
  const int bm = blockIdx.y * 128, bn = blockIdx.x * 128;

  const int arow = tid >> 3;           // 0..31
  const int ac   = (tid & 7) * 4;      // 0..28
  const int bkr  = tid >> 5;           // 0..7
  const int bnc  = (tid & 31) * 4;     // 0..124

  v8f acc[2][4] = {};

  for (int kt = 0; kt < K; kt += 32) {
    __syncthreads();
    #pragma unroll
    for (int i = 0; i < 4; ++i) {                       // A tile 128x32
      int r = arow + i * 32;
      float4 f = *(const float4*)&A[(size_t)(bm + r) * K + kt + ac];
      ushort4 p;
      p.x = f32_to_bf16(f.x); p.y = f32_to_bf16(f.y);
      p.z = f32_to_bf16(f.z); p.w = f32_to_bf16(f.w);
      *(ushort4*)&lA[r][ac] = p;
    }
    #pragma unroll
    for (int i = 0; i < 4; ++i) {                       // B tile 32x128 -> [n][k]
      int k = bkr + i * 8;
      float4 f = *(const float4*)&Bw[(size_t)(kt + k) * N + bn + bnc];
      lB[bnc + 0][k] = f32_to_bf16(f.x);
      lB[bnc + 1][k] = f32_to_bf16(f.y);
      lB[bnc + 2][k] = f32_to_bf16(f.z);
      lB[bnc + 3][k] = f32_to_bf16(f.w);
    }
    if (kt + 32 < K) {                                  // global_prefetch_b8
      __builtin_prefetch(&A[(size_t)(bm + arow) * K + kt + 32 + ac], 0, 1);
      __builtin_prefetch(&Bw[(size_t)(kt + 32 + bkr) * N + bn + bnc], 0, 1);
    }
    __syncthreads();

    Frag aF[2];
    #pragma unroll
    for (int i = 0; i < 2; ++i) {                       // A frags: 16x32 each
      int row = wm * 32 + i * 16 + lm;
      aF[i].h[0] = *(const v8us*)&lA[row][half * 16];
      aF[i].h[1] = *(const v8us*)&lA[row][half * 16 + 8];
    }
    #pragma unroll
    for (int j = 0; j < 4; ++j) {                       // B frags: 32x16 each
      int col = wn * 64 + j * 16 + lm;
      v16us bF = *(const v16us*)&lB[col][half * 16];
      #pragma unroll
      for (int i = 0; i < 2; ++i)
        acc[i][j] = wmma_bf16(aF[i].v, bF, acc[i][j]);
    }
  }

  // Epilogue: C element (r,lane) -> row = base + r + 8*half, col = base + lm.
  #pragma unroll
  for (int i = 0; i < 2; ++i) {
    int gmb = bm + wm * 32 + i * 16 + half * 8;
    #pragma unroll
    for (int j = 0; j < 4; ++j) {
      int gn = bn + wn * 64 + j * 16 + lm;
      float bv = bias[gn];
      #pragma unroll
      for (int r = 0; r < 8; ++r) {
        int gm = gmb + r;
        float val = acc[i][j][r] + bv;
        int bb = gm >> 10, s = gm & 1023;
        if (gn < HID) {                                  // Q: [b,h,s,d] bf16
          int h = gn >> 7, d = gn & 127;
          wq[((size_t)(bb * NH + h) * SS + s) * HD + d] = f32_to_bf16(val);
        } else if (gn < 2 * HID) {                       // K: [b,h,s,d] + cache
          int c = gn - HID, h = c >> 7, d = c & 127;
          wk[((size_t)(bb * NH + h) * SS + s) * HD + d] = f32_to_bf16(val);
          kcache[(size_t)slots[gm] * HID + c] = val;
        } else {                                         // V: [b,h,d,s] + cache
          int c = gn - 2 * HID, h = c >> 7, d = c & 127;
          wv[((size_t)(bb * NH + h) * HD + d) * SS + s] = f32_to_bf16(val);
          vcache[(size_t)slots[gm] * HID + c] = val;
        }
      }
    }
  }
}

// ---------------------------------------------------------------------------
// Causal flash attention. Block = 128 q rows (8 waves x 16), K-tile = 64.
// K/V tiles staged into LDS with async b128 copies (ASYNCcnt path).
// ---------------------------------------------------------------------------
__global__ __launch_bounds__(256) void attn_kernel(
    const unsigned short* __restrict__ wq,
    const unsigned short* __restrict__ wk,
    const unsigned short* __restrict__ wv,
    unsigned short* __restrict__ wattn)
{
  constexpr int KT = 64;
  __shared__ __align__(64) unsigned short lK[KT][HD];    // [krow][d]
  __shared__ __align__(64) unsigned short lV[HD][KT];    // [d][krow]
  __shared__ __align__(64) unsigned short lP[128][KT];   // [qrow][krow]

  const int tid  = threadIdx.x;
  const int lane = tid & 31, wave = tid >> 5;
  const int half = lane >> 4, lm = lane & 15;
  const int qt = blockIdx.x, h = blockIdx.y, b = blockIdx.z;
  const int sbase = qt * 128;
  const int s0 = sbase + wave * 16;                       // wave's q rows

  const unsigned short* qb = wq + ((size_t)(b * NH + h) * SS + s0) * HD;
  const unsigned short* kb = wk + ((size_t)(b * NH + h) * SS) * HD;
  const unsigned short* vb = wv + ((size_t)(b * NH + h) * HD) * SS;

  Frag aQ[4];
  {
    const unsigned short* qr = qb + (size_t)lm * HD;
    #pragma unroll
    for (int kc = 0; kc < 4; ++kc) {
      aQ[kc].h[0] = *(const v8us*)&qr[kc * 32 + half * 16];
      aQ[kc].h[1] = *(const v8us*)&qr[kc * 32 + half * 16 + 8];
    }
  }

  float mi[8], li[8];
  #pragma unroll
  for (int r = 0; r < 8; ++r) { mi[r] = -1e30f; li[r] = 0.f; }
  v8f oacc[8] = {};
  const float scale = 0.08838834764831845f;               // 1/sqrt(128)

  const int nkt = qt * 2 + 2;                             // causal tile bound
  for (int kt = 0; kt < nkt; ++kt) {
    __syncthreads();
    #pragma unroll
    for (int i = 0; i < 4; ++i) {                         // lK: 64x128 bf16
      int c = tid + i * 256;
      int kr = c >> 4, cc = (c & 15) * 8;
      async_copy_b128(&kb[(size_t)(kt * KT + kr) * HD + cc], &lK[kr][cc]);
    }
    #pragma unroll
    for (int i = 0; i < 4; ++i) {                         // lV: [d][krow]
      int c = tid + i * 256;
      int d = c >> 3, cc = (c & 7) * 8;
      async_copy_b128(&vb[(size_t)d * SS + kt * KT + cc], &lV[d][cc]);
    }
    async_wait();
    __syncthreads();

    if (kt * KT <= s0 + 15) {                             // wave-uniform skip
      // ---- scores S = Q K^T ----
      v8f sacc[4] = {};
      #pragma unroll
      for (int nt = 0; nt < 4; ++nt) {
        #pragma unroll
        for (int kc = 0; kc < 4; ++kc) {
          v16us bF = *(const v16us*)&lK[nt * 16 + lm][kc * 32 + half * 16];
          sacc[nt] = wmma_bf16(aQ[kc].v, bF, sacc[nt]);
        }
      }
      // ---- causal mask + online softmax ----
      float rmax[8];
      #pragma unroll
      for (int r = 0; r < 8; ++r) rmax[r] = -1e30f;
      #pragma unroll
      for (int nt = 0; nt < 4; ++nt) {
        int col = kt * KT + nt * 16 + lm;
        #pragma unroll
        for (int r = 0; r < 8; ++r) {
          int row = s0 + half * 8 + r;
          float v = sacc[nt][r] * scale;
          if (col > row) v = -1e30f;
          sacc[nt][r] = v;
          rmax[r] = fmaxf(rmax[r], v);
        }
      }
      #pragma unroll
      for (int r = 0; r < 8; ++r) {                       // 16-lane reduce
        #pragma unroll
        for (int m = 1; m < 16; m <<= 1)
          rmax[r] = fmaxf(rmax[r], __shfl_xor(rmax[r], m, 32));
      }
      float alpha[8], rsum[8];
      #pragma unroll
      for (int r = 0; r < 8; ++r) {
        float mn = fmaxf(mi[r], rmax[r]);
        alpha[r] = __expf(mi[r] - mn);
        mi[r] = mn;
        rsum[r] = 0.f;
      }
      #pragma unroll
      for (int nt = 0; nt < 4; ++nt) {
        #pragma unroll
        for (int r = 0; r < 8; ++r) {
          float p = __expf(sacc[nt][r] - mi[r]);
          sacc[nt][r] = p;
          rsum[r] += p;
        }
      }
      #pragma unroll
      for (int r = 0; r < 8; ++r) {
        #pragma unroll
        for (int m = 1; m < 16; m <<= 1)
          rsum[r] += __shfl_xor(rsum[r], m, 32);
        li[r] = li[r] * alpha[r] + rsum[r];
      }
      #pragma unroll
      for (int nt = 0; nt < 8; ++nt)
        #pragma unroll
        for (int r = 0; r < 8; ++r)
          oacc[nt][r] *= alpha[r];
      // ---- P via LDS: C-layout -> A-fragment layout ----
      #pragma unroll
      for (int nt = 0; nt < 4; ++nt)
        #pragma unroll
        for (int r = 0; r < 8; ++r)
          lP[wave * 16 + half * 8 + r][nt * 16 + lm] = f32_to_bf16(sacc[nt][r]);
      Frag aP[2];
      #pragma unroll
      for (int kc = 0; kc < 2; ++kc) {
        aP[kc].h[0] = *(const v8us*)&lP[wave * 16 + lm][kc * 32 + half * 16];
        aP[kc].h[1] = *(const v8us*)&lP[wave * 16 + lm][kc * 32 + half * 16 + 8];
      }
      // ---- O += P V ----
      #pragma unroll
      for (int nt = 0; nt < 8; ++nt) {
        #pragma unroll
        for (int kc = 0; kc < 2; ++kc) {
          v16us bF = *(const v16us*)&lV[nt * 16 + lm][kc * 32 + half * 16];
          oacc[nt] = wmma_bf16(aP[kc].v, bF, oacc[nt]);
        }
      }
    }
  }

  float rli[8];
  #pragma unroll
  for (int r = 0; r < 8; ++r) rli[r] = 1.0f / li[r];
  #pragma unroll
  for (int nt = 0; nt < 8; ++nt) {
    #pragma unroll
    for (int r = 0; r < 8; ++r) {
      int t = b * SS + s0 + half * 8 + r;
      int col = h * HD + nt * 16 + lm;
      wattn[(size_t)t * HID + col] = f32_to_bf16(oacc[nt][r] * rli[r]);
    }
  }
}

// ---------------------------------------------------------------------------
// Output projection: out[4096,2048] = attn[4096,2048] @ o_w[2048,2048] + o_b
// A is already bf16 -> staged to LDS with async b128 copies.
// ---------------------------------------------------------------------------
__global__ __launch_bounds__(256) void oproj_gemm_kernel(
    const unsigned short* __restrict__ A, const float* __restrict__ Bw,
    const float* __restrict__ bias, float* __restrict__ out)
{
  const int K = HID, N = HID;
  __shared__ __align__(64) unsigned short lA[128][32];
  __shared__ __align__(64) unsigned short lB[128][32];

  const int tid  = threadIdx.x;
  const int lane = tid & 31, wave = tid >> 5;
  const int wm = wave >> 1, wn = wave & 1;
  const int half = lane >> 4, lm = lane & 15;
  const int bm = blockIdx.y * 128, bn = blockIdx.x * 128;

  const int bkr  = tid >> 5;
  const int bnc  = (tid & 31) * 4;

  v8f acc[2][4] = {};

  for (int kt = 0; kt < K; kt += 32) {
    __syncthreads();
    #pragma unroll
    for (int i = 0; i < 2; ++i) {                        // A already bf16: async
      int c = tid + i * 256;
      int r = c >> 2, cc = (c & 3) * 8;
      async_copy_b128(&A[(size_t)(bm + r) * K + kt + cc], &lA[r][cc]);
    }
    #pragma unroll
    for (int i = 0; i < 4; ++i) {
      int k = bkr + i * 8;
      float4 f = *(const float4*)&Bw[(size_t)(kt + k) * N + bn + bnc];
      lB[bnc + 0][k] = f32_to_bf16(f.x);
      lB[bnc + 1][k] = f32_to_bf16(f.y);
      lB[bnc + 2][k] = f32_to_bf16(f.z);
      lB[bnc + 3][k] = f32_to_bf16(f.w);
    }
    if (kt + 32 < K) {
      __builtin_prefetch(&Bw[(size_t)(kt + 32 + bkr) * N + bn + bnc], 0, 1);
    }
    async_wait();
    __syncthreads();

    Frag aF[2];
    #pragma unroll
    for (int i = 0; i < 2; ++i) {
      int row = wm * 32 + i * 16 + lm;
      aF[i].h[0] = *(const v8us*)&lA[row][half * 16];
      aF[i].h[1] = *(const v8us*)&lA[row][half * 16 + 8];
    }
    #pragma unroll
    for (int j = 0; j < 4; ++j) {
      int col = wn * 64 + j * 16 + lm;
      v16us bF = *(const v16us*)&lB[col][half * 16];
      #pragma unroll
      for (int i = 0; i < 2; ++i)
        acc[i][j] = wmma_bf16(aF[i].v, bF, acc[i][j]);
    }
  }

  #pragma unroll
  for (int i = 0; i < 2; ++i) {
    int gmb = bm + wm * 32 + i * 16 + half * 8;
    #pragma unroll
    for (int j = 0; j < 4; ++j) {
      int gn = bn + wn * 64 + j * 16 + lm;
      float bv = bias[gn];
      #pragma unroll
      for (int r = 0; r < 8; ++r)
        out[(size_t)(gmb + r) * N + gn] = acc[i][j][r] + bv;
    }
  }
}

// ---------------------------------------------------------------------------
extern "C" void kernel_launch(void* const* d_in, const int* in_sizes, int n_in,
                              void* d_out, int out_size, void* d_ws, size_t ws_size,
                              hipStream_t stream) {
  (void)in_sizes; (void)n_in; (void)out_size; (void)ws_size;
  const float* hs     = (const float*)d_in[0];   // [4096,2048]
  const float* qkv_w  = (const float*)d_in[1];   // [2048,6144]
  const float* qkv_b  = (const float*)d_in[2];   // [6144]
  const float* o_w    = (const float*)d_in[3];   // [2048,2048]
  const float* o_b    = (const float*)d_in[4];   // [2048]
  const int*   slots  = (const int*)d_in[8];     // [4096]

  float* out    = (float*)d_out;                               // [4096,2048]
  float* kcache = out + (size_t)TT * HID;                      // [8192,16,128]
  float* vcache = kcache + (size_t)NSLOT * NH * HD;            // [8192,16,128]

  unsigned short* wq    = (unsigned short*)d_ws;               // bf16 [b,h,s,d]
  unsigned short* wk    = wq + (size_t)TT * HID;               // bf16 [b,h,s,d]
  unsigned short* wv    = wk + (size_t)TT * HID;               // bf16 [b,h,d,s]
  unsigned short* wattn = wv + (size_t)TT * HID;               // bf16 [t,hid]

  // 1) zero both cache regions (reference caches start as zeros)
  long n4 = (long)2 * NSLOT * NH * HD / 4;
  zero_kernel<<<4096, 256, 0, stream>>>(kcache, n4);

  // 2) fused QKV projection + reshape_and_cache scatter
  qkv_gemm_kernel<<<dim3(48, 32), 256, 0, stream>>>(
      hs, qkv_w, qkv_b, slots, wq, wk, wv, kcache, vcache);

  // 3) causal flash attention
  attn_kernel<<<dim3(8, NH, NB), 256, 0, stream>>>(wq, wk, wv, wattn);

  // 4) output projection
  oproj_gemm_kernel<<<dim3(16, 32), 256, 0, stream>>>(wattn, o_w, o_b, out);
}
